// SCNN_21792664060497
// MI455X (gfx1250) — matrix operand order repaired and checked by reading, not compiled, not run on hardware
//
#include <hip/hip_runtime.h>
#include <hip/hip_bf16.h>

typedef __attribute__((ext_vector_type(16))) __bf16 v16bf;
typedef __attribute__((ext_vector_type(8)))  float  v8f;

#define C_DIM 128
#define H_DIM 36
#define W_DIM 100
#define KTAP  9
#define NK_STEPS 36            // 9 taps * (128 ci / 32 per wmma)
#define STRIDE 144             // LDS row stride in bf16 elements (bank-shifted)
#define PER_PASS (8 * 36 * 32 * 16)   // bf16 elements of repacked weights per pass

// --- explicit global-address-space accessors: force global_load/store (not
// --- flat_*, which would couple LOADcnt with DScnt and serialize vs LDS waits)
__device__ __forceinline__ v16bf load_a_global(const __bf16* p) {
    return *(const __attribute__((address_space(1))) v16bf*)(const v16bf*)p;
}
__device__ __forceinline__ float load_f_global(const float* p) {
    return *(const __attribute__((address_space(1))) float*)p;
}
__device__ __forceinline__ void store_f_global(float* p, float v) {
    *(__attribute__((address_space(1))) float*)p = v;
}

// ---------------------------------------------------------------------------
// Repack f32 [co=128][ci=128][tap=9] weights into bf16 WMMA A-fragment layout:
// ws[p][mi(8 co-tiles)][s(36 k-steps)][lane(32)][j(16)]
// K ordering: K = tap*128 + ci. Contiguous 16-bit A 16x32 layout:
//   lanes 0-15  (M=lane):    j0..15 -> K 0..15
//   lanes 16-31 (M=lane-16): j0..15 -> K 16..31
// ---------------------------------------------------------------------------
__global__ void scnn_repack_weights(const float* __restrict__ w0,
                                    const float* __restrict__ w1,
                                    const float* __restrict__ w2,
                                    const float* __restrict__ w3,
                                    __bf16* __restrict__ wsW) {
    int gid = blockIdx.x * blockDim.x + threadIdx.x;
    if (gid >= 4 * PER_PASS) return;
    int p   = gid / PER_PASS;
    int rem = gid % PER_PASS;
    int j    = rem & 15;
    int lane = (rem >> 4) & 31;
    int s    = (rem >> 9) % 36;
    int mi   = rem / (36 * 32 * 16);
    const float* src = (p == 0) ? w0 : (p == 1) ? w1 : (p == 2) ? w2 : w3;
    int M    = mi * 16 + (lane & 15);
    int half = lane >> 4;
    int Klocal = j + 16 * half;
    int tap = s >> 2;
    int ci  = ((s & 3) << 5) + Klocal;
    wsW[gid] = (__bf16)load_f_global(&src[((size_t)M * C_DIM + ci) * KTAP + tap]);
}

// ---------------------------------------------------------------------------
// One directional pass. VERT: scan over H, conv along W (nlen=100, 7 N-tiles);
// !VERT: scan over W, conv along H (nlen=36, 3 N-tiles).
// state: LDS bf16 shadow of previous output row, [n+4][ci] with stride STRIDE.
// Recurrent f32 state lives in `out` (global, L2-resident); in-place safe.
// ---------------------------------------------------------------------------
template <int NTILES, bool VERT>
__device__ void scnn_pass(const float* __restrict__ in, float* __restrict__ out,
                          const __bf16* __restrict__ wA, __bf16* state,
                          int b, bool rev) {
    const int tid  = threadIdx.x;
    const int lane = tid & 31;
    const int wave = tid >> 5;          // wave32: 8 waves, one co-tile each
    const int half = lane >> 4;
    const int ncol = lane & 15;
    const int nsteps = VERT ? H_DIM : W_DIM;
    const int nlen   = VERT ? W_DIM : H_DIM;

    // clear LDS state (zero padding regions + stale data)
    {
        unsigned* st32 = (unsigned*)state;
        for (int idx = tid; idx < (C_DIM * STRIDE) / 2; idx += 256) st32[idx] = 0u;
    }
    __syncthreads();

    const __bf16* wBase = wA + ((size_t)wave * NK_STEPS * 32 + lane) * 16;

    for (int st = 0; st < nsteps; ++st) {
        const int i = rev ? (nsteps - 1 - st) : st;

        if (st == 0) {
            // first row: out = x, state = bf16(x)
            for (int idx = tid; idx < C_DIM * nlen; idx += 256) {
                int c = idx / nlen, n = idx % nlen;
                int hh = VERT ? i : n, ww = VERT ? n : i;
                size_t g = (((size_t)b * C_DIM + c) * H_DIM + hh) * W_DIM + ww;
                float v = load_f_global(&in[g]);
                store_f_global(&out[g], v);
                state[(n + 4) * STRIDE + c] = (__bf16)v;
            }
            __syncthreads();
            continue;
        }

        // prefetch this step's additive input row (L2-resident)
        {
            int hh = VERT ? i : ncol, ww = VERT ? ncol : i;
            size_t g = (((size_t)b * C_DIM + wave * 16) * H_DIM + hh) * W_DIM + ww;
            __builtin_prefetch(&in[g], 0, 3);
        }

        // y = Wgt x prev_row : M=128, K=1152, N=nlen (padded to NTILES*16)
        v8f acc[NTILES];
        #pragma unroll
        for (int t = 0; t < NTILES; ++t)
            acc[t] = (v8f){0.f, 0.f, 0.f, 0.f, 0.f, 0.f, 0.f, 0.f};

        // k-loop: batch all NTILES B-fragment LDS loads, then issue the
        // NTILES independent WMMAs back-to-back (one s_wait per k-step, and
        // unroll-4 lets the scheduler pipeline next step's loads under WMMA).
        #pragma unroll 4
        for (int s = 0; s < NK_STEPS; ++s) {
            const v16bf a = load_a_global(wBase + (size_t)s * 32 * 16);
            const int tap = s >> 2;
            const int cib = ((s & 3) << 5) + (half << 4);
            v16bf bm[NTILES];
            #pragma unroll
            for (int t = 0; t < NTILES; ++t)
                bm[t] = *(const v16bf*)&state[(t * 16 + ncol + tap) * STRIDE + cib];
            #pragma unroll
            for (int t = 0; t < NTILES; ++t)
                acc[t] = __builtin_amdgcn_wmma_f32_16x16x32_bf16(
                    false, a, false, bm[t], (short)0, acc[t], false, false);
        }

        __syncthreads();   // all reads of prev state done before overwrite

        // out = in + relu(y); refresh bf16 LDS state
        #pragma unroll
        for (int t = 0; t < NTILES; ++t) {
            int n = t * 16 + ncol;
            if (n < nlen) {
                #pragma unroll
                for (int r = 0; r < 8; ++r) {
                    int co = wave * 16 + r + (half << 3);
                    int hh = VERT ? i : n, ww = VERT ? n : i;
                    size_t g = (((size_t)b * C_DIM + co) * H_DIM + hh) * W_DIM + ww;
                    float y = acc[t][r];
                    float v = load_f_global(&in[g]) + (y > 0.f ? y : 0.f);
                    store_f_global(&out[g], v);
                    state[(n + 4) * STRIDE + co] = (__bf16)v;
                }
            }
        }
        __syncthreads();   // new state visible before next step's WMMA
    }
}

// ---------------------------------------------------------------------------
// One workgroup per image: runs all 4 passes back-to-back (dependencies are
// strictly per-image, so no global sync is ever needed).
// ---------------------------------------------------------------------------
__global__ void __launch_bounds__(256, 1)
scnn_main(const float* __restrict__ x, float* __restrict__ out,
          const __bf16* __restrict__ wsW) {
    __shared__ __bf16 state[C_DIM * STRIDE];   // 36.9 KB of 320 KB WGP LDS
    const int b = blockIdx.x;
    scnn_pass<7, true >(x,   out, wsW + 0 * (size_t)PER_PASS, state, b, false);
    scnn_pass<7, true >(out, out, wsW + 1 * (size_t)PER_PASS, state, b, true );
    scnn_pass<3, false>(out, out, wsW + 2 * (size_t)PER_PASS, state, b, false);
    scnn_pass<3, false>(out, out, wsW + 3 * (size_t)PER_PASS, state, b, true );
}

extern "C" void kernel_launch(void* const* d_in, const int* in_sizes, int n_in,
                              void* d_out, int out_size, void* d_ws, size_t ws_size,
                              hipStream_t stream) {
    const float* x    = (const float*)d_in[0];
    const float* w_ud = (const float*)d_in[1];
    const float* w_du = (const float*)d_in[2];
    const float* w_lr = (const float*)d_in[3];
    const float* w_rl = (const float*)d_in[4];
    float* out  = (float*)d_out;
    __bf16* wsW = (__bf16*)d_ws;   // needs 4*PER_PASS*2 = 1.18 MB scratch

    const int total = 4 * PER_PASS;
    scnn_repack_weights<<<(total + 255) / 256, 256, 0, stream>>>(
        w_ud, w_du, w_lr, w_rl, wsW);

    scnn_main<<<32, 256, 0, stream>>>(x, out, wsW);
}
